// Mamba_80083960201315
// MI455X (gfx1250) — matrix operand order repaired and checked by reading, not compiled, not run on hardware
//
#include <hip/hip_runtime.h>
#include <hip/hip_bf16.h>
#include <math.h>

// ---------------- problem constants ----------------
#define N_LAYERS 2
#define D_CONV   4
#define D_MODEL  768
#define D_INNER  1536
#define D_STATE  16
#define D_DELTA  48
#define BATCH    2
#define SEQ      1024
#define ROWS     (BATCH * SEQ)      // 2048
#define D_XZ     (2 * D_INNER)      // 3072
#define N_XP     (D_DELTA + 2 * D_STATE)  // 80
#define N_XP_PAD 96                 // padded to 3 WMMA n-tiles
#define K_DT_PAD 64                 // delta K 48 -> 64
#define NCHUNK   16                 // parallel-scan chunks
#define CLEN     (SEQ / NCHUNK)     // 64
#define DBLK     (D_INNER / 256)    // 6

typedef __bf16 bf16;
typedef __attribute__((ext_vector_type(16))) __bf16 v16bf;
typedef __attribute__((ext_vector_type(8)))  __bf16 v8bf;
typedef __attribute__((ext_vector_type(8)))  float  v8f;

__device__ __forceinline__ v8f wmma_bf16(v16bf a, v16bf b, v8f c) {
  return __builtin_amdgcn_wmma_f32_16x16x32_bf16(false, a, false, b, (short)0, c,
                                                 false, false);
}

// ---------------- generic WMMA GEMM: C[MxN] = A[MxK] * W[NxK]^T (+ epilogue) --------
// Fully compile-time shapes/strides so the backend emits base+imm-offset clauses.
// Wave tile: 32 x (16*NT).  EPILOG: 0 = none, 1 = +bias then softplus, 2 = +residual
template <int NT, int EPILOG, int M, int N, int K, int LDA, int LDW, int LDC>
__global__ void __launch_bounds__(256) wmma_gemm_kernel(
    const bf16* __restrict__ A, const bf16* __restrict__ W, float* __restrict__ C,
    const float* __restrict__ bias, const float* __restrict__ resid) {
  constexpr int TILES_N = N / (16 * NT);
  constexpr int TILES   = (M / 32) * TILES_N;
  const int lane = threadIdx.x & 31;
  const int wave = threadIdx.x >> 5;
  const int tile = blockIdx.x * 8 + wave;    // one 32x(16*NT) tile per wave
  if (tile >= TILES) return;
  const int m0  = (tile / TILES_N) * 32;
  const int n0  = (tile % TILES_N) * (16 * NT);
  const int loc = lane & 15;
  const int hi  = lane >> 4;

  // Per-lane base pointers (this kernel's contiguous-k order, same for A and B):
  // A 16x32: lane m=loc, K = hi*16+0..15 contiguous
  // B 32x16 from W[N][K] row-major: lane n=loc holds K = hi*16+0..15 contiguous
  const bf16* Ap = A + (size_t)(m0 + loc) * LDA + hi * 16;
  const bf16* Wp = W + (size_t)(n0 + loc) * LDW + hi * 16;

  v8f acc[2][NT] = {};
  for (int k0 = 0; k0 < K; k0 += 32) {
    v16bf a0, a1, b[NT];
    {
      const v8bf c0 = *(const v8bf*)(Ap + k0);
      const v8bf c1 = *(const v8bf*)(Ap + k0 + 8);
      const v8bf d0 = *(const v8bf*)(Ap + 16 * LDA + k0);
      const v8bf d1 = *(const v8bf*)(Ap + 16 * LDA + k0 + 8);
#pragma unroll
      for (int i = 0; i < 8; ++i) {
        a0[i] = c0[i]; a0[i + 8] = c1[i];
        a1[i] = d0[i]; a1[i + 8] = d1[i];
      }
    }
#pragma unroll
    for (int j = 0; j < NT; ++j)
      b[j] = *(const v16bf*)(Wp + (size_t)j * 16 * LDW + k0);
#pragma unroll
    for (int j = 0; j < NT; ++j) {
      acc[0][j] = wmma_bf16(a0, b[j], acc[0][j]);
      acc[1][j] = wmma_bf16(a1, b[j], acc[1][j]);
    }
  }
#pragma unroll
  for (int mi = 0; mi < 2; ++mi)
#pragma unroll
    for (int nj = 0; nj < NT; ++nj) {
      const int n = n0 + nj * 16 + loc;
#pragma unroll
      for (int r = 0; r < 8; ++r) {
        const int m = m0 + mi * 16 + hi * 8 + r;   // C/D: lanes16-31 -> M+8
        float v = acc[mi][nj][r];
        if (EPILOG == 1) { v += bias[n]; v = (v > 20.f) ? v : log1pf(__expf(v)); }
        if (EPILOG == 2) { v += resid[(size_t)m * LDC + n]; }
        C[(size_t)m * LDC + n] = v;
      }
    }
}

// ---------------- RMSNorm -> bf16 ----------------
__global__ void __launch_bounds__(256) rmsnorm_kernel(const float* __restrict__ x,
                                                      const float* __restrict__ w,
                                                      bf16* __restrict__ out) {
  __shared__ float sm[256];
  const int row = blockIdx.x, tid = threadIdx.x;
  const float* xr = x + (size_t)row * D_MODEL;
  float v0 = xr[tid], v1 = xr[tid + 256], v2 = xr[tid + 512];
  sm[tid] = v0 * v0 + v1 * v1 + v2 * v2;
  __syncthreads();
  for (int off = 128; off > 0; off >>= 1) {
    if (tid < off) sm[tid] += sm[tid + off];
    __syncthreads();
  }
  const float scale = rsqrtf(sm[0] / (float)D_MODEL + 1e-5f);
  bf16* o = out + (size_t)row * D_MODEL;
  o[tid]       = (bf16)(w[tid] * v0 * scale);
  o[tid + 256] = (bf16)(w[tid + 256] * v1 * scale);
  o[tid + 512] = (bf16)(w[tid + 512] * v2 * scale);
}

// ---------------- causal depthwise conv (K=4) + SiLU ----------------
__global__ void conv_silu_kernel(const float* __restrict__ xz,
                                 const float* __restrict__ cw,
                                 const float* __restrict__ cb,
                                 float* __restrict__ xc_f, bf16* __restrict__ xc_h) {
  const int idx = blockIdx.x * blockDim.x + threadIdx.x;
  if (idx >= ROWS * D_INNER) return;
  const int d    = idx % D_INNER;
  const int bl   = idx / D_INNER;
  const int l    = bl % SEQ;
  const int base = bl - l;  // b*SEQ
  float acc = cb[d];
#pragma unroll
  for (int j = 0; j < D_CONV; ++j) {
    const int ll = l - (D_CONV - 1) + j;
    if (ll >= 0) acc += cw[d * D_CONV + j] * xz[(size_t)(base + ll) * D_XZ + d];
  }
  const float s = acc / (1.f + __expf(-acc));   // silu
  xc_f[(size_t)bl * D_INNER + d] = s;
  xc_h[(size_t)bl * D_INNER + d] = (bf16)s;
}

// ---------------- weight / activation conversion helpers ----------------
__global__ void cvt_bf16_kernel(const float* __restrict__ in, bf16* __restrict__ out, int n) {
  const int i = blockIdx.x * blockDim.x + threadIdx.x;
  if (i < n) out[i] = (bf16)in[i];
}

__global__ void cvt_xproj_kernel(const float* __restrict__ in, bf16* __restrict__ out) {
  const int i = blockIdx.x * blockDim.x + threadIdx.x;
  if (i >= N_LAYERS * N_XP_PAD * D_INNER) return;
  const int layer = i / (N_XP_PAD * D_INNER);
  const int r = i % (N_XP_PAD * D_INNER);
  const int n = r / D_INNER, k = r % D_INNER;
  out[i] = (n < N_XP) ? (bf16)in[(size_t)layer * N_XP * D_INNER + (size_t)n * D_INNER + k]
                      : (bf16)0.f;
}

__global__ void cvt_dt_kernel(const float* __restrict__ in, bf16* __restrict__ out) {
  const int i = blockIdx.x * blockDim.x + threadIdx.x;
  if (i >= N_LAYERS * D_INNER * K_DT_PAD) return;
  const int layer = i / (D_INNER * K_DT_PAD);
  const int r = i % (D_INNER * K_DT_PAD);
  const int n = r / K_DT_PAD, k = r % K_DT_PAD;
  out[i] = (k < D_DELTA) ? (bf16)in[(size_t)layer * D_INNER * D_DELTA + (size_t)n * D_DELTA + k]
                         : (bf16)0.f;
}

__global__ void prep_delta_kernel(const float* __restrict__ dbc, bf16* __restrict__ out) {
  const int i = blockIdx.x * blockDim.x + threadIdx.x;
  if (i >= ROWS * K_DT_PAD) return;
  const int row = i / K_DT_PAD, k = i % K_DT_PAD;
  out[i] = (k < D_DELTA) ? (bf16)dbc[(size_t)row * N_XP_PAD + k] : (bf16)0.f;
}

// ---------------- chunked parallel selective-scan ----------------
// h_t = dA_t * h_{t-1} + dBx_t is diagonal-linear; chunk composes as
// h_end = (prod dA) * h_start + h_local.  Pass A: per-chunk prod & local state.
// Pass B: tiny boundary scan.  Pass C: replay with corrected h0 + fused output path.

__global__ void __launch_bounds__(256) scanA_kernel(
    const float* __restrict__ delta, const float* __restrict__ dbc,
    const float* __restrict__ xc, const float* __restrict__ A_log,
    float* __restrict__ Pprod, float* __restrict__ hloc) {
  __shared__ float sB[CLEN * D_STATE];   // 4 KB
  const int blk = blockIdx.x;
  const int b = blk / (NCHUNK * DBLK);
  const int r = blk % (NCHUNK * DBLK);
  const int c = r / DBLK;
  const int d = (r % DBLK) * 256 + threadIdx.x;
  const int row0 = b * SEQ + c * CLEN;
  for (int i = threadIdx.x; i < CLEN * D_STATE; i += 256) {
    const int l = i >> 4, n = i & 15;
    sB[i] = dbc[(size_t)(row0 + l) * N_XP_PAD + D_DELTA + n];
  }
  __syncthreads();
  float a[D_STATE], h[D_STATE], P[D_STATE];
#pragma unroll
  for (int n = 0; n < D_STATE; ++n) {
    a[n] = -__expf(A_log[(size_t)d * D_STATE + n]);
    h[n] = 0.f;
    P[n] = 1.f;
  }
#pragma unroll 4
  for (int l = 0; l < CLEN; ++l) {
    const size_t row = (size_t)(row0 + l);
    const float dl = delta[row * D_INNER + d];
    const float xv = xc[row * D_INNER + d];
    const float dx = dl * xv;
#pragma unroll
    for (int n = 0; n < D_STATE; ++n) {
      const float dA = __expf(dl * a[n]);
      P[n] *= dA;
      h[n] = dA * h[n] + dx * sB[l * D_STATE + n];
    }
  }
  const size_t o = (((size_t)b * NCHUNK + c) * D_INNER + d) * D_STATE;
#pragma unroll
  for (int n = 0; n < D_STATE; ++n) { Pprod[o + n] = P[n]; hloc[o + n] = h[n]; }
}

__global__ void __launch_bounds__(256) scanB_kernel(
    const float* __restrict__ Pprod, const float* __restrict__ hloc,
    float* __restrict__ Hinit) {
  const int idx = blockIdx.x * 256 + threadIdx.x;   // < BATCH * D_INNER * D_STATE
  const int b  = idx / (D_INNER * D_STATE);
  const int rn = idx % (D_INNER * D_STATE);
  float H = 0.f;
#pragma unroll
  for (int c = 0; c < NCHUNK; ++c) {
    const size_t o = ((size_t)(b * NCHUNK + c) * D_INNER * D_STATE) + rn;
    Hinit[o] = H;
    H = Pprod[o] * H + hloc[o];
  }
}

__global__ void __launch_bounds__(256) scanC_kernel(
    const float* __restrict__ delta, const float* __restrict__ dbc,
    const float* __restrict__ xc, const float* __restrict__ xz,
    const float* __restrict__ A_log, const float* __restrict__ Dp,
    const float* __restrict__ Hinit, bf16* __restrict__ y_out) {
  __shared__ float sBC[CLEN * 2 * D_STATE];   // 8 KB
  const int blk = blockIdx.x;
  const int b = blk / (NCHUNK * DBLK);
  const int r = blk % (NCHUNK * DBLK);
  const int c = r / DBLK;
  const int d = (r % DBLK) * 256 + threadIdx.x;
  const int row0 = b * SEQ + c * CLEN;
  for (int i = threadIdx.x; i < CLEN * 2 * D_STATE; i += 256) {
    const int l = i >> 5, j = i & 31;
    sBC[i] = dbc[(size_t)(row0 + l) * N_XP_PAD + D_DELTA + j];
  }
  __syncthreads();
  float a[D_STATE], h[D_STATE];
  const size_t o = (((size_t)b * NCHUNK + c) * D_INNER + d) * D_STATE;
#pragma unroll
  for (int n = 0; n < D_STATE; ++n) {
    a[n] = -__expf(A_log[(size_t)d * D_STATE + n]);
    h[n] = Hinit[o + n];
  }
  const float Dd = Dp[d];
#pragma unroll 2
  for (int l = 0; l < CLEN; ++l) {
    const size_t row = (size_t)(row0 + l);
    const float dl = delta[row * D_INNER + d];
    const float xv = xc[row * D_INNER + d];
    const float zv = xz[row * D_XZ + D_INNER + d];
    const float dx = dl * xv;
    float y = 0.f;
#pragma unroll
    for (int n = 0; n < D_STATE; ++n) {
      const float dA = __expf(dl * a[n]);
      h[n] = dA * h[n] + dx * sBC[l * 2 * D_STATE + n];
      y += h[n] * sBC[l * 2 * D_STATE + D_STATE + n];
    }
    const float g = zv / (1.f + __expf(-zv));
    y_out[row * D_INNER + d] = (bf16)((y + Dd * xv) * g);
  }
}

// ---------------- host orchestration ----------------
extern "C" void kernel_launch(void* const* d_in, const int* in_sizes, int n_in,
                              void* d_out, int out_size, void* d_ws, size_t ws_size,
                              hipStream_t stream) {
  (void)in_sizes; (void)n_in; (void)out_size; (void)ws_size;
  const float* x_in      = (const float*)d_in[0];
  const float* in_proj_w = (const float*)d_in[1];
  const float* conv_w    = (const float*)d_in[2];
  const float* conv_b    = (const float*)d_in[3];
  const float* xproj_w   = (const float*)d_in[4];
  const float* dt_w      = (const float*)d_in[5];
  const float* dt_b      = (const float*)d_in[6];
  const float* A_log     = (const float*)d_in[7];
  const float* Dvec      = (const float*)d_in[8];
  const float* out_proj_w= (const float*)d_in[9];
  const float* norm_w    = (const float*)d_in[10];

  char* ws = (char*)d_ws;
  size_t off = 0;
  auto alloc = [&](size_t bytes) {
    char* p = ws + off;
    off = (off + bytes + 255) & ~(size_t)255;
    return p;
  };
  bf16*  win   = (bf16*)alloc((size_t)N_LAYERS * D_XZ * D_MODEL * 2);
  bf16*  wout  = (bf16*)alloc((size_t)N_LAYERS * D_MODEL * D_INNER * 2);
  bf16*  wxp   = (bf16*)alloc((size_t)N_LAYERS * N_XP_PAD * D_INNER * 2);
  bf16*  wdt   = (bf16*)alloc((size_t)N_LAYERS * D_INNER * K_DT_PAD * 2);
  float* xcur  = (float*)alloc((size_t)ROWS * D_MODEL * 4);
  bf16*  xn    = (bf16*) alloc((size_t)ROWS * D_MODEL * 2);
  float* xz    = (float*)alloc((size_t)ROWS * D_XZ * 4);
  float* xcf   = (float*)alloc((size_t)ROWS * D_INNER * 4);
  bf16*  xch   = (bf16*) alloc((size_t)ROWS * D_INNER * 2);
  float* dbc   = (float*)alloc((size_t)ROWS * N_XP_PAD * 4);
  bf16*  dpad  = (bf16*) alloc((size_t)ROWS * K_DT_PAD * 2);
  float* delta = (float*)alloc((size_t)ROWS * D_INNER * 4);
  bf16*  ybf   = (bf16*) alloc((size_t)ROWS * D_INNER * 2);
  float* Pprod = (float*)alloc((size_t)BATCH * NCHUNK * D_INNER * D_STATE * 4);
  float* hloc  = (float*)alloc((size_t)BATCH * NCHUNK * D_INNER * D_STATE * 4);
  float* Hinit = (float*)alloc((size_t)BATCH * NCHUNK * D_INNER * D_STATE * 4);

  // running residual
  hipMemcpyAsync(xcur, x_in, (size_t)ROWS * D_MODEL * sizeof(float),
                 hipMemcpyDeviceToDevice, stream);

  // one-time weight conversions (both layers)
  {
    int n1 = N_LAYERS * D_XZ * D_MODEL;
    cvt_bf16_kernel<<<(n1 + 255) / 256, 256, 0, stream>>>(in_proj_w, win, n1);
    int n2 = N_LAYERS * D_MODEL * D_INNER;
    cvt_bf16_kernel<<<(n2 + 255) / 256, 256, 0, stream>>>(out_proj_w, wout, n2);
    int n3 = N_LAYERS * N_XP_PAD * D_INNER;
    cvt_xproj_kernel<<<(n3 + 255) / 256, 256, 0, stream>>>(xproj_w, wxp);
    int n4 = N_LAYERS * D_INNER * K_DT_PAD;
    cvt_dt_kernel<<<(n4 + 255) / 256, 256, 0, stream>>>(dt_w, wdt);
  }

  for (int layer = 0; layer < N_LAYERS; ++layer) {
    rmsnorm_kernel<<<ROWS, 256, 0, stream>>>(xcur, norm_w + layer * D_MODEL, xn);

    // in_proj: [2048x768] x [3072x768]^T -> xz [2048x3072]   (wave tile 32x64)
    wmma_gemm_kernel<4, 0, ROWS, D_XZ, D_MODEL, D_MODEL, D_MODEL, D_XZ>
        <<<(64 * 48) / 8, 256, 0, stream>>>(
        xn, win + (size_t)layer * D_XZ * D_MODEL, xz, nullptr, nullptr);

    conv_silu_kernel<<<(ROWS * D_INNER) / 256, 256, 0, stream>>>(
        xz, conv_w + (size_t)layer * D_INNER * D_CONV, conv_b + (size_t)layer * D_INNER,
        xcf, xch);

    // xproj: [2048x1536] x [96x1536]^T -> dbc [2048x96]      (wave tile 32x32)
    wmma_gemm_kernel<2, 0, ROWS, N_XP_PAD, D_INNER, D_INNER, D_INNER, N_XP_PAD>
        <<<(64 * 3) / 8, 256, 0, stream>>>(
        xch, wxp + (size_t)layer * N_XP_PAD * D_INNER, dbc, nullptr, nullptr);

    prep_delta_kernel<<<(ROWS * K_DT_PAD) / 256, 256, 0, stream>>>(dbc, dpad);

    // delta proj + bias + softplus: [2048x64] x [1536x64]^T -> delta [2048x1536]
    wmma_gemm_kernel<4, 1, ROWS, D_INNER, K_DT_PAD, K_DT_PAD, K_DT_PAD, D_INNER>
        <<<(64 * 24) / 8, 256, 0, stream>>>(
        dpad, wdt + (size_t)layer * D_INNER * K_DT_PAD, delta,
        dt_b + (size_t)layer * D_INNER, nullptr);

    // chunked parallel scan (3 passes)
    scanA_kernel<<<BATCH * NCHUNK * DBLK, 256, 0, stream>>>(
        delta, dbc, xcf, A_log + (size_t)layer * D_INNER * D_STATE, Pprod, hloc);
    scanB_kernel<<<(BATCH * D_INNER * D_STATE) / 256, 256, 0, stream>>>(
        Pprod, hloc, Hinit);
    scanC_kernel<<<BATCH * NCHUNK * DBLK, 256, 0, stream>>>(
        delta, dbc, xcf, xz,
        A_log + (size_t)layer * D_INNER * D_STATE, Dvec + (size_t)layer * D_INNER,
        Hinit, ybf);

    // out_proj + residual: [2048x1536] x [768x1536]^T (+ xcur) -> next x
    float* outp = (layer == N_LAYERS - 1) ? (float*)d_out : xcur;
    wmma_gemm_kernel<4, 2, ROWS, D_MODEL, D_INNER, D_INNER, D_INNER, D_MODEL>
        <<<(64 * 12) / 8, 256, 0, stream>>>(
        ybf, wout + (size_t)layer * D_MODEL * D_INNER, outp, nullptr, xcur);
  }
}
